// LyapunovThinkingBlock_90941637526290
// MI455X (gfx1250) — matrix operand order repaired and checked by reading, not compiled, not run on hardware
//
#include <hip/hip_runtime.h>
#include <hip/hip_bf16.h>

#define D       896
#define BTILE   32          // rows per workgroup (2 WMMA M-tiles)
#define NTILES  56          // D / 16
#define KSTEPS  28          // D / 32
#define BATCH   32768

typedef __attribute__((ext_vector_type(16))) __bf16         v16bf;
typedef __attribute__((ext_vector_type(8)))  float          v8f;
typedef __attribute__((ext_vector_type(8)))  unsigned short ushort8;

union FragBF {
  v16bf   v;
  ushort8 u[2];
};

__device__ __forceinline__ unsigned short f2bf(float f) {
  unsigned u = __builtin_bit_cast(unsigned, f);
  u += 0x7FFFu + ((u >> 16) & 1u);          // round-to-nearest-even
  return (unsigned short)(u >> 16);
}

__device__ __forceinline__ float silu_f(float x) {
  return x * (1.0f / (1.0f + __expf(-x)));
}

// A-matrix 16x32 bf16 fragment from an LDS tile (row-major, row stride D).
// lanes 0-15 -> M=lane, K={kb..kb+7, kb+16..kb+23}, kb=0; lanes 16-31 -> kb=8.
__device__ __forceinline__ v16bf load_a(const unsigned short* tile, int k_off, int lane) {
  const int m  = lane & 15;
  const int kb = (lane < 16) ? 0 : 8;
  const unsigned short* p = tile + m * D + k_off + kb;
  FragBF f;
  f.u[0] = *(const ushort8*)(p);
  f.u[1] = *(const ushort8*)(p + 16);
  return f.v;
}

// B-matrix 32x16 bf16 fragment: B[k][n] = W[n][k] (W row-major, row stride ldk).
__device__ __forceinline__ v16bf load_b(const unsigned short* w, int ldk, int n,
                                        int k_off, int lane) {
  const int kb = (lane < 16) ? 0 : 8;
  const unsigned short* p = w + (size_t)n * ldk + k_off + kb;
  FragBF f;
  f.u[0] = *(const ushort8*)(p);
  f.u[1] = *(const ushort8*)(p + 16);
  return f.v;
}

#define WMMA_BF16(a, b, acc) \
  __builtin_amdgcn_wmma_f32_16x16x32_bf16(false, (a), false, (b), (short)0, (acc), false, false)

__device__ __forceinline__ void init_acc(v8f& a, float bias) {
#pragma unroll
  for (int i = 0; i < 8; ++i) a[i] = bias;
}

// 2M x 2N block: 2 A-frags + 2 B-frags -> 4 independent WMMAs per k-step.
__device__ __forceinline__ void mm_pair(v8f& c00, v8f& c10, v8f& c01, v8f& c11,
                                        const unsigned short* tile,
                                        const unsigned short* w, int ldk, int kw0,
                                        int n0, int n1, int lane) {
  for (int kt = 0; kt < KSTEPS; ++kt) {
    const int k = kt * 32;
    v16bf b0 = load_b(w, ldk, n0, kw0 + k, lane);
    v16bf b1 = load_b(w, ldk, n1, kw0 + k, lane);
    v16bf a0 = load_a(tile,          k, lane);
    v16bf a1 = load_a(tile + 16 * D, k, lane);
    c00 = WMMA_BF16(a0, b0, c00);
    c10 = WMMA_BF16(a1, b0, c10);
    c01 = WMMA_BF16(a0, b1, c01);
    c11 = WMMA_BF16(a1, b1, c11);
  }
}

// 2M x 1N tail tile.
__device__ __forceinline__ void mm_one(v8f& c0, v8f& c1,
                                       const unsigned short* tile,
                                       const unsigned short* w, int ldk, int kw0,
                                       int n, int lane) {
  for (int kt = 0; kt < KSTEPS; ++kt) {
    const int k = kt * 32;
    v16bf b  = load_b(w, ldk, n, kw0 + k, lane);
    v16bf a0 = load_a(tile,          k, lane);
    v16bf a1 = load_a(tile + 16 * D, k, lane);
    c0 = WMMA_BF16(a0, b, c0);
    c1 = WMMA_BF16(a1, b, c1);
  }
}

__device__ __forceinline__ void store_bf16(unsigned short* dst, int n, int hi8,
                                           const v8f& c0, const v8f& c1) {
#pragma unroll
  for (int r = 0; r < 8; ++r) {
    dst[(r + hi8) * D + n]      = f2bf(c0[r]);
    dst[(r + hi8 + 16) * D + n] = f2bf(c1[r]);
  }
}

__device__ __forceinline__ void store_silu(unsigned short* dst, int n, int hi8,
                                           const v8f& c0, const v8f& c1) {
#pragma unroll
  for (int r = 0; r < 8; ++r) {
    dst[(r + hi8) * D + n]      = f2bf(silu_f(c0[r]));
    dst[(r + hi8 + 16) * D + n] = f2bf(silu_f(c1[r]));
  }
}

__device__ __forceinline__ void store_f32(float* dst, int n, int hi8,
                                          const v8f& c0, const v8f& c1) {
#pragma unroll
  for (int r = 0; r < 8; ++r) {
    dst[(r + hi8) * D + n]      = c0[r];
    dst[(r + hi8 + 16) * D + n] = c1[r];
  }
}

__global__ void convert_f32_to_bf16(const float* __restrict__ src,
                                    unsigned short* __restrict__ dst, int n) {
  int i = blockIdx.x * blockDim.x + threadIdx.x;
  if (i < n) dst[i] = f2bf(src[i]);
}

__global__ __launch_bounds__(256, 1) void fused_lyapunov(
    const float* __restrict__ h_t, const float* __restrict__ phi_x,
    const unsigned short* __restrict__ wv_bf, const unsigned short* __restrict__ wo_bf,
    const unsigned short* __restrict__ w1_bf, const unsigned short* __restrict__ w2_bf,
    const float* __restrict__ bv, const float* __restrict__ bo,
    const float* __restrict__ b1, const float* __restrict__ b2,
    const float* __restrict__ ln_g, const float* __restrict__ ln_b,
    float* __restrict__ out) {
  // 224 KB LDS, hand-aliased (320 KB WGP budget, 1 WG/WGP):
  //   [0,       57344)  sHt    bf16 [32][896]   (live whole kernel)
  //   [57344,  114688)  sPhiG1 bf16 [32][896]   (phi_x, reused for g1)
  //   [114688, 229376)  sG2    f32  [32][896]   (overlays sV + sHatt)
  //   [114688, 172032)  sV     bf16 [32][896]   (dead after GEMM2)
  //   [172032, 229376)  sHatt  bf16 [32][896]   (dead after GEMM3)
  __shared__ __align__(16) unsigned char smem[BTILE * D * 2 * 2 + BTILE * D * 4];
  unsigned short* sHt    = (unsigned short*)smem;
  unsigned short* sPhiG1 = (unsigned short*)(smem + BTILE * D * 2);
  unsigned short* sV     = (unsigned short*)(smem + BTILE * D * 4);
  unsigned short* sHatt  = (unsigned short*)(smem + BTILE * D * 4 + BTILE * D * 2);
  float*          sG2    = (float*)(smem + BTILE * D * 4);

  const int tid  = threadIdx.x;
  const int lane = tid & 31;
  const int wave = tid >> 5;
  const int row0 = blockIdx.x * BTILE;

  // ---- stage inputs -> LDS (bf16), vectorized 16-elem chunks ----
  {
    const int r  = tid >> 3;          // 0..31 rows
    const int j  = tid & 7;           // 8 threads per row, 7 chunks each
    const float* hp = h_t   + (size_t)(row0 + r) * D;
    const float* pp = phi_x + (size_t)(row0 + r) * D;
    for (int ch = j; ch < NTILES; ch += 8) {
      const int c = ch * 16;
      ushort8 lo, hi;
#pragma unroll
      for (int e = 0; e < 8; ++e) {
        lo[e] = f2bf(hp[c + e]);
        hi[e] = f2bf(hp[c + 8 + e]);
      }
      *(ushort8*)(sHt + r * D + c)     = lo;
      *(ushort8*)(sHt + r * D + c + 8) = hi;
#pragma unroll
      for (int e = 0; e < 8; ++e) {
        lo[e] = f2bf(pp[c + e]);
        hi[e] = f2bf(pp[c + 8 + e]);
      }
      *(ushort8*)(sPhiG1 + r * D + c)     = lo;
      *(ushort8*)(sPhiG1 + r * D + c + 8) = hi;
    }
  }
  __syncthreads();

  const int nlo = lane & 15;
  const int hi8 = (lane < 16) ? 0 : 8;
  const int ntail = (wave + 48) * 16 + nlo;

  // ---- GEMM1: v = phi_x @ Wv^T + bv  -> sV ----
#pragma unroll 1
  for (int p = 0; p < 3; ++p) {
    const int n0 = (wave + 16 * p) * 16 + nlo;
    const int n1 = n0 + 128;
    v8f c00, c10, c01, c11;
    init_acc(c00, bv[n0]); init_acc(c10, bv[n0]);
    init_acc(c01, bv[n1]); init_acc(c11, bv[n1]);
    mm_pair(c00, c10, c01, c11, sPhiG1, wv_bf, D, 0, n0, n1, lane);
    store_bf16(sV, n0, hi8, c00, c10);
    store_bf16(sV, n1, hi8, c01, c11);
  }
  {
    v8f c0, c1;
    init_acc(c0, bv[ntail]); init_acc(c1, bv[ntail]);
    mm_one(c0, c1, sPhiG1, wv_bf, D, 0, ntail, lane);
    store_bf16(sV, ntail, hi8, c0, c1);
  }
  __syncthreads();

  // ---- GEMM2: h_att = v @ Wo^T + bo  -> sHatt ----
#pragma unroll 1
  for (int p = 0; p < 3; ++p) {
    const int n0 = (wave + 16 * p) * 16 + nlo;
    const int n1 = n0 + 128;
    v8f c00, c10, c01, c11;
    init_acc(c00, bo[n0]); init_acc(c10, bo[n0]);
    init_acc(c01, bo[n1]); init_acc(c11, bo[n1]);
    mm_pair(c00, c10, c01, c11, sV, wo_bf, D, 0, n0, n1, lane);
    store_bf16(sHatt, n0, hi8, c00, c10);
    store_bf16(sHatt, n1, hi8, c01, c11);
  }
  {
    v8f c0, c1;
    init_acc(c0, bo[ntail]); init_acc(c1, bo[ntail]);
    mm_one(c0, c1, sV, wo_bf, D, 0, ntail, lane);
    store_bf16(sHatt, ntail, hi8, c0, c1);
  }
  __syncthreads();

  // ---- GEMM3: g1 = silu([h_t, h_att] @ W1^T + b1) -> sPhiG1 (phi dead) ----
#pragma unroll 1
  for (int p = 0; p < 3; ++p) {
    const int n0 = (wave + 16 * p) * 16 + nlo;
    const int n1 = n0 + 128;
    v8f c00, c10, c01, c11;
    init_acc(c00, b1[n0]); init_acc(c10, b1[n0]);
    init_acc(c01, b1[n1]); init_acc(c11, b1[n1]);
    mm_pair(c00, c10, c01, c11, sHt,   w1_bf, 2 * D, 0, n0, n1, lane);  // K: h_t
    mm_pair(c00, c10, c01, c11, sHatt, w1_bf, 2 * D, D, n0, n1, lane);  // K: h_att
    store_silu(sPhiG1, n0, hi8, c00, c10);
    store_silu(sPhiG1, n1, hi8, c01, c11);
  }
  {
    v8f c0, c1;
    init_acc(c0, b1[ntail]); init_acc(c1, b1[ntail]);
    mm_one(c0, c1, sHt,   w1_bf, 2 * D, 0, ntail, lane);
    mm_one(c0, c1, sHatt, w1_bf, 2 * D, D, ntail, lane);
    store_silu(sPhiG1, ntail, hi8, c0, c1);
  }
  __syncthreads();

  // ---- GEMM4: g2 = g1 @ W2^T + b2 -> sG2 (f32, overlays dead sV/sHatt) ----
#pragma unroll 1
  for (int p = 0; p < 3; ++p) {
    const int n0 = (wave + 16 * p) * 16 + nlo;
    const int n1 = n0 + 128;
    v8f c00, c10, c01, c11;
    init_acc(c00, b2[n0]); init_acc(c10, b2[n0]);
    init_acc(c01, b2[n1]); init_acc(c11, b2[n1]);
    mm_pair(c00, c10, c01, c11, sPhiG1, w2_bf, D, 0, n0, n1, lane);
    store_f32(sG2, n0, hi8, c00, c10);
    store_f32(sG2, n1, hi8, c01, c11);
  }
  {
    v8f c0, c1;
    init_acc(c0, b2[ntail]); init_acc(c1, b2[ntail]);
    mm_one(c0, c1, sPhiG1, w2_bf, D, 0, ntail, lane);
    store_f32(sG2, ntail, hi8, c0, c1);
  }
  __syncthreads();

  // ---- LayerNorm + residual: out = h_t + (g2 - mu)*rstd*g + b ----
  {
    const int r = tid >> 3;           // 0..31 rows, 8 lanes per row
    const int j = tid & 7;
    float sum = 0.f, sq = 0.f;
    for (int c = j; c < D; c += 8) {
      const float g = sG2[r * D + c];
      sum += g;
      sq  += g * g;
    }
#pragma unroll
    for (int off = 4; off; off >>= 1) {
      sum += __shfl_xor(sum, off, 8);
      sq  += __shfl_xor(sq, off, 8);
    }
    const float mu   = sum * (1.0f / D);
    const float var  = sq * (1.0f / D) - mu * mu;
    const float rstd = rsqrtf(var + 1e-5f);
    const float* hp = h_t + (size_t)(row0 + r) * D;   // f32 residual from L2
    float* op = out + (size_t)(row0 + r) * D;
    for (int c = j; c < D; c += 8) {
      const float g = (sG2[r * D + c] - mu) * rstd;
      op[c] = hp[c] + g * ln_g[c] + ln_b[c];
    }
  }
}

extern "C" void kernel_launch(void* const* d_in, const int* in_sizes, int n_in,
                              void* d_out, int out_size, void* d_ws, size_t ws_size,
                              hipStream_t stream) {
  (void)in_sizes; (void)n_in; (void)out_size; (void)ws_size;
  const float* h_t        = (const float*)d_in[0];
  const float* phi_x      = (const float*)d_in[1];
  const float* in_proj_w  = (const float*)d_in[2];
  const float* in_proj_b  = (const float*)d_in[3];
  const float* out_proj_w = (const float*)d_in[4];
  const float* out_proj_b = (const float*)d_in[5];
  const float* w1         = (const float*)d_in[6];
  const float* b1         = (const float*)d_in[7];
  const float* w2         = (const float*)d_in[8];
  const float* b2         = (const float*)d_in[9];
  const float* ln_g       = (const float*)d_in[10];
  const float* ln_b       = (const float*)d_in[11];
  float* out = (float*)d_out;

  // bf16 weight copies in workspace (q/k projections are dead: softmax over len-1 == 1)
  unsigned short* wv_bf = (unsigned short*)d_ws;        // [D][D]
  unsigned short* wo_bf = wv_bf + (size_t)D * D;        // [D][D]
  unsigned short* w1_bf = wo_bf + (size_t)D * D;        // [D][2D]
  unsigned short* w2_bf = w1_bf + (size_t)D * 2 * D;    // [D][D]

  const int ndd = D * D;
  const int n2d = D * 2 * D;
  convert_f32_to_bf16<<<(ndd + 255) / 256, 256, 0, stream>>>(in_proj_w + (size_t)2 * D * D, wv_bf, ndd);
  convert_f32_to_bf16<<<(ndd + 255) / 256, 256, 0, stream>>>(out_proj_w, wo_bf, ndd);
  convert_f32_to_bf16<<<(n2d + 255) / 256, 256, 0, stream>>>(w1, w1_bf, n2d);
  convert_f32_to_bf16<<<(ndd + 255) / 256, 256, 0, stream>>>(w2, w2_bf, ndd);

  fused_lyapunov<<<BATCH / BTILE, 256, 0, stream>>>(
      h_t, phi_x, wv_bf, wo_bf, w1_bf, w2_bf,
      in_proj_b + (size_t)2 * D, out_proj_b, b1, b2, ln_g, ln_b, out);
}
